// TriangleMultiplication_19112604467460
// MI455X (gfx1250) — matrix operand order repaired and checked by reading, not compiled, not run on hardware
//
#include <hip/hip_runtime.h>

#define NDIM 512
#define CDIM 128
#define NNSQ (NDIM*NDIM)

typedef __bf16 bf16;
typedef __attribute__((ext_vector_type(16))) __bf16 v16bf;
typedef __attribute__((ext_vector_type(8)))  __bf16 v8bf;
typedef __attribute__((ext_vector_type(8)))  float  v8f;

// A-fragment (16-bit A 16x32): lane holds K = klo..klo+7 and klo+16..klo+23
__device__ __forceinline__ v16bf frag_a(const bf16* p) {
  v8bf lo = *(const v8bf*)(p);
  v8bf hi = *(const v8bf*)(p + 16);
  return __builtin_shufflevector(lo, hi, 0,1,2,3,4,5,6,7,8,9,10,11,12,13,14,15);
}
// B-fragment (16-bit B 32x16): lane holds K = klo..klo+7 and klo+16..klo+23
__device__ __forceinline__ v16bf frag_b(const bf16* p) {
  v8bf lo = *(const v8bf*)(p);
  v8bf hi = *(const v8bf*)(p + 16);
  return __builtin_shufflevector(lo, hi, 0,1,2,3,4,5,6,7,8,9,10,11,12,13,14,15);
}
__device__ __forceinline__ v8f wmma_bf16(v16bf a, v16bf b, v8f c) {
  return __builtin_amdgcn_wmma_f32_16x16x32_bf16(false, a, false, b, (short)0, c, false, false);
}
__device__ __forceinline__ float sigm(float x) { return 1.0f / (1.0f + __expf(-x)); }

// ---------------------------------------------------------------------------
// Prep: W^T bf16.  WcatT[640][128]: rows [ap|ag|bp|bg|g] columns, k-contiguous.
// WzT[128][128]: WzT[cz][k] = W_z[k][cz].
// ---------------------------------------------------------------------------
__global__ __launch_bounds__(256)
void tm_prep(const float* __restrict__ Wap, const float* __restrict__ Wag,
             const float* __restrict__ Wbp, const float* __restrict__ Wbg,
             const float* __restrict__ Wg,  const float* __restrict__ Wz,
             bf16* __restrict__ WcatT, bf16* __restrict__ WzT)
{
  int idx = blockIdx.x * blockDim.x + threadIdx.x;
  if (idx < 5 * CDIM * CDIM) {
    int seg = idx / (CDIM * CDIM);
    int r   = idx % (CDIM * CDIM);
    int j = r >> 7, k = r & 127;
    const float* W = (seg == 0) ? Wap : (seg == 1) ? Wag : (seg == 2) ? Wbp
                   : (seg == 3) ? Wbg : Wg;
    WcatT[(size_t)(seg * CDIM + j) * CDIM + k] = (bf16)W[k * CDIM + j];
  } else if (idx < 6 * CDIM * CDIM) {
    int r = idx - 5 * CDIM * CDIM;
    int j = r >> 7, k = r & 127;
    WzT[(size_t)j * CDIM + k] = (bf16)Wz[k * CDIM + j];
  }
}

// ---------------------------------------------------------------------------
// Stage 1: LayerNorm(z) -> 5 projections via WMMA -> gated a,b (bf16,
// channel-major) and g (fp32, row-major).  Block: 64 rows (fixed i, 64 k).
// ---------------------------------------------------------------------------
__global__ __launch_bounds__(128)
void tm_stage1(const float* __restrict__ z,
               const float* __restrict__ lng, const float* __restrict__ lnb,
               const bf16* __restrict__ WcatT,
               const float* __restrict__ bap, const float* __restrict__ bag,
               const float* __restrict__ bbp, const float* __restrict__ bbg,
               const float* __restrict__ bgg,
               bf16* __restrict__ a_t, bf16* __restrict__ b_t,
               float* __restrict__ gbuf)
{
  __shared__ __align__(16) char smem[51200];
  float* zsh  = (float*)smem;               // [64][132] raw z rows (phase 1)
  bf16*  absh = (bf16*)smem;                // [2][128ch][64k] staging (phase 3+)
  bf16*  znsh = (bf16*)(smem + 33792);      // [64][136] bf16 normalized rows

  const int tid  = threadIdx.x;
  const int lane = tid & 31;
  const int wave = tid >> 5;
  const int n    = lane & 15;
  const int lh   = (lane >> 4) & 1;
  const int i    = blockIdx.x >> 3;
  const int k0   = (blockIdx.x & 7) * 64;
  const int r0   = i * NDIM + k0;           // first flat z row

  // load 64 rows of z
  for (int idx = tid; idx < 2048; idx += 128) {
    int row = idx >> 5, q = idx & 31;
    *(float4*)&zsh[row * 132 + q * 4] =
        *(const float4*)&z[(size_t)(r0 + row) * CDIM + q * 4];
  }
  __syncthreads();

  // fp32 LayerNorm per row -> bf16
  if (tid < 64) {
    const float* xr = &zsh[tid * 132];
    float s = 0.f, s2 = 0.f;
    for (int c = 0; c < CDIM; ++c) { float v = xr[c]; s += v; s2 += v * v; }
    float mu  = s * (1.0f / CDIM);
    float var = s2 * (1.0f / CDIM) - mu * mu;
    float rs  = rsqrtf(var + 1e-5f);
    bf16* zr = &znsh[tid * 136];
    for (int c = 0; c < CDIM; ++c)
      zr[c] = (bf16)((xr[c] - mu) * rs * lng[c] + lnb[c]);
  }
  __syncthreads();

  const int mrow0 = wave * 16;              // this wave's 16 k-rows
  for (int cg = 0; cg < 8; ++cg) {          // 16-channel output groups
    const int col = cg * 16 + n;
    v8f aAP = {}, aAG = {}, aBP = {}, aBG = {}, aG = {};
    for (int kt = 0; kt < 4; ++kt) {
      v16bf af = frag_a(&znsh[(mrow0 + n) * 136 + kt * 32 + lh * 8]);
      const bf16* wb = &WcatT[(size_t)col * CDIM + kt * 32 + lh * 8];
      aAP = wmma_bf16(af, frag_b(wb + (size_t)0 * CDIM * CDIM), aAP);
      aAG = wmma_bf16(af, frag_b(wb + (size_t)1 * CDIM * CDIM), aAG);
      aBP = wmma_bf16(af, frag_b(wb + (size_t)2 * CDIM * CDIM), aBP);
      aBG = wmma_bf16(af, frag_b(wb + (size_t)3 * CDIM * CDIM), aBG);
      aG  = wmma_bf16(af, frag_b(wb + (size_t)4 * CDIM * CDIM), aG);
    }
    float vap = bap[col], vag = bag[col], vbp = bbp[col], vbg = bbg[col],
          vg = bgg[col];
    for (int v = 0; v < 8; ++v) {
      int kl = mrow0 + v + lh * 8;
      float av = sigm(aAG[v] + vag) * (aAP[v] + vap);
      float bv = sigm(aBG[v] + vbg) * (aBP[v] + vbp);
      float gv = sigm(aG[v] + vg);
      absh[col * 64 + kl]             = (bf16)av;
      absh[128 * 64 + col * 64 + kl]  = (bf16)bv;
      gbuf[(size_t)(r0 + kl) * CDIM + col] = gv;   // 16 consecutive lanes -> 64B
    }
  }
  __syncthreads();

  // channel-major writes: 64 consecutive k per channel = 128B contiguous
  const size_t plane = (size_t)i * NDIM + k0;
  for (int idx = tid; idx < 8192; idx += 128) {
    int ch = idx >> 6, kk = idx & 63;
    size_t off = (size_t)ch * NNSQ + plane + kk;
    a_t[off] = absh[idx];
    b_t[off] = absh[8192 + idx];
  }
}

// ---------------------------------------------------------------------------
// Stage 2: per-channel tri_c = A_c * B_c^T.  Block: one channel, 128x128 tile.
// ---------------------------------------------------------------------------
__global__ __launch_bounds__(256)
void tm_stage2(const bf16* __restrict__ a_t, const bf16* __restrict__ b_t,
               float* __restrict__ tri_t)
{
  __shared__ __align__(16) char smem[32768];
  bf16*  At  = (bf16*)smem;             // [128][40]
  bf16*  Bt  = (bf16*)(smem + 10240);   // [128][40]
  float* Tst = (float*)smem;            // later: [64][128] output staging

  const int tid = threadIdx.x, lane = tid & 31, wave = tid >> 5;
  const int n = lane & 15, lh = (lane >> 4) & 1;
  const int c  = blockIdx.x >> 4;
  const int tt = blockIdx.x & 15;
  const int i0 = (tt >> 2) * 128, j0 = (tt & 3) * 128;
  const int wi = wave & 3, wj = wave >> 2;
  const int ri0 = wi * 32, rj0 = wj * 64;     // wave computes 32x64 region
  const size_t base = (size_t)c * NNSQ;

  v8f acc[2][4] = {};

  for (int kc = 0; kc < 16; ++kc) {
    const int kk0 = kc * 32;
    __syncthreads();
    for (int idx = tid; idx < 512; idx += 256) {
      int row = idx >> 2, q = idx & 3;
      *(uint4*)&At[row * 40 + q * 8] =
          *(const uint4*)&a_t[base + (size_t)(i0 + row) * NDIM + kk0 + q * 8];
      *(uint4*)&Bt[row * 40 + q * 8] =
          *(const uint4*)&b_t[base + (size_t)(j0 + row) * NDIM + kk0 + q * 8];
    }
    __syncthreads();
    v16bf af0 = frag_a(&At[(ri0 +  0 + n) * 40 + lh * 8]);
    v16bf af1 = frag_a(&At[(ri0 + 16 + n) * 40 + lh * 8]);
    v16bf bf0 = frag_b(&Bt[(rj0 +  0 + n) * 40 + lh * 8]);
    v16bf bf1 = frag_b(&Bt[(rj0 + 16 + n) * 40 + lh * 8]);
    v16bf bf2 = frag_b(&Bt[(rj0 + 32 + n) * 40 + lh * 8]);
    v16bf bf3 = frag_b(&Bt[(rj0 + 48 + n) * 40 + lh * 8]);
    acc[0][0] = wmma_bf16(af0, bf0, acc[0][0]);
    acc[0][1] = wmma_bf16(af0, bf1, acc[0][1]);
    acc[0][2] = wmma_bf16(af0, bf2, acc[0][2]);
    acc[0][3] = wmma_bf16(af0, bf3, acc[0][3]);
    acc[1][0] = wmma_bf16(af1, bf0, acc[1][0]);
    acc[1][1] = wmma_bf16(af1, bf1, acc[1][1]);
    acc[1][2] = wmma_bf16(af1, bf2, acc[1][2]);
    acc[1][3] = wmma_bf16(af1, bf3, acc[1][3]);
  }
  __syncthreads();

  // stage + coalesced write, two 64-row halves (reuses tile LDS)
  for (int half = 0; half < 2; ++half) {
    if ((wi >> 1) == half) {
      for (int s = 0; s < 2; ++s)
        for (int t = 0; t < 4; ++t)
          for (int v = 0; v < 8; ++v) {
            int il = ri0 - half * 64 + s * 16 + v + lh * 8;
            int jl = rj0 + t * 16 + n;
            Tst[il * 128 + jl] = acc[s][t][v];
          }
    }
    __syncthreads();
    for (int idx = tid; idx < 8192; idx += 256) {
      int il = idx >> 7, jl = idx & 127;
      tri_t[base + (size_t)(i0 + half * 64 + il) * NDIM + (j0 + jl)] = Tst[idx];
    }
    __syncthreads();
  }
}

// ---------------------------------------------------------------------------
// Stage 3: transpose-read tri (128ch x 64pos), fp32 LN over channels,
// bf16 WMMA with Wz^T, fuse g * (. + b_z), coalesced fp32 out.
// ---------------------------------------------------------------------------
__global__ __launch_bounds__(256)
void tm_stage3(const float* __restrict__ tri_t, const float* __restrict__ gbuf,
               const float* __restrict__ lng, const float* __restrict__ lnb,
               const bf16* __restrict__ WzT, const float* __restrict__ bz,
               float* __restrict__ out)
{
  __shared__ __align__(16) char smem[50176];
  float* Tsh = (float*)smem;             // [128c][64p]; later [64p][128cz]
  bf16*  nA  = (bf16*)(smem + 32768);    // [64][136] normalized bf16

  const int tid = threadIdx.x, lane = tid & 31, wave = tid >> 5;
  const int n = lane & 15, lh = (lane >> 4) & 1;
  const int p0 = blockIdx.x * 64;

  for (int idx = tid; idx < 2048; idx += 256) {
    int cc = idx >> 4, q = idx & 15;
    *(float4*)&Tsh[cc * 64 + q * 4] =
        *(const float4*)&tri_t[(size_t)cc * NNSQ + p0 + q * 4];
  }
  __syncthreads();

  if (tid < 64) {
    int p = tid;
    float s = 0.f, s2 = 0.f;
    for (int cc = 0; cc < CDIM; ++cc) {
      float v = Tsh[cc * 64 + p]; s += v; s2 += v * v;
    }
    float mu  = s * (1.0f / CDIM);
    float var = s2 * (1.0f / CDIM) - mu * mu;
    float rs  = rsqrtf(var + 1e-5f);
    for (int cc = 0; cc < CDIM; ++cc)
      nA[p * 136 + cc] = (bf16)((Tsh[cc * 64 + p] - mu) * rs * lng[cc] + lnb[cc]);
  }
  __syncthreads();

  const int prow0 = (wave & 3) * 16;
  const int jt0   = (wave >> 2) * 4;
  v16bf af[4];
  for (int kt = 0; kt < 4; ++kt)
    af[kt] = frag_a(&nA[(prow0 + n) * 136 + kt * 32 + lh * 8]);
  v8f acc[4];
  for (int jj = 0; jj < 4; ++jj) {
    v8f a_ = {};
    const bf16* wb = &WzT[(size_t)((jt0 + jj) * 16 + n) * CDIM + lh * 8];
    for (int kt = 0; kt < 4; ++kt)
      a_ = wmma_bf16(af[kt], frag_b(wb + kt * 32), a_);
    acc[jj] = a_;
  }
  __syncthreads();
  for (int jj = 0; jj < 4; ++jj)
    for (int v = 0; v < 8; ++v) {
      int pl = prow0 + v + lh * 8;
      Tsh[pl * 128 + (jt0 + jj) * 16 + n] = acc[jj][v];
    }
  __syncthreads();
  for (int idx = tid; idx < 8192; idx += 256) {
    int pl = idx >> 7, cz = idx & 127;
    size_t gp = (size_t)(p0 + pl) * CDIM + cz;
    out[gp] = gbuf[gp] * (Tsh[idx] + bz[cz]);
  }
}

// ---------------------------------------------------------------------------
extern "C" void kernel_launch(void* const* d_in, const int* in_sizes, int n_in,
                              void* d_out, int out_size, void* d_ws, size_t ws_size,
                              hipStream_t stream)
{
  (void)in_sizes; (void)n_in; (void)out_size; (void)ws_size;
  const float* z    = (const float*)d_in[0];
  const float* lnig = (const float*)d_in[1];
  const float* lnib = (const float*)d_in[2];
  const float* Wap  = (const float*)d_in[3];
  const float* bap  = (const float*)d_in[4];
  const float* Wag  = (const float*)d_in[5];
  const float* bag  = (const float*)d_in[6];
  const float* Wbp  = (const float*)d_in[7];
  const float* bbp  = (const float*)d_in[8];
  const float* Wbg  = (const float*)d_in[9];
  const float* bbg  = (const float*)d_in[10];
  const float* Wg   = (const float*)d_in[11];
  const float* bg   = (const float*)d_in[12];
  const float* lnog = (const float*)d_in[13];
  const float* lnob = (const float*)d_in[14];
  const float* Wz   = (const float*)d_in[15];
  const float* bz   = (const float*)d_in[16];

  // workspace layout (~403 MB total)
  char* ws = (char*)d_ws;
  size_t off = 0;
  auto carve = [&](size_t bytes) -> char* {
    char* p = ws + off;
    off += (bytes + 255) & ~(size_t)255;
    return p;
  };
  bf16*  a_t   = (bf16*) carve((size_t)CDIM * NNSQ * 2);   //  64 MB
  bf16*  b_t   = (bf16*) carve((size_t)CDIM * NNSQ * 2);   //  64 MB
  float* gbuf  = (float*)carve((size_t)NNSQ * CDIM * 4);   // 128 MB
  float* tri   = (float*)carve((size_t)CDIM * NNSQ * 4);   // 128 MB
  bf16*  WcatT = (bf16*) carve((size_t)5 * CDIM * CDIM * 2);
  bf16*  WzT   = (bf16*) carve((size_t)CDIM * CDIM * 2);

  tm_prep<<<(6 * CDIM * CDIM + 255) / 256, 256, 0, stream>>>(
      Wap, Wag, Wbp, Wbg, Wg, Wz, WcatT, WzT);
  tm_stage1<<<NDIM * (NDIM / 64), 128, 0, stream>>>(
      z, lnig, lnib, WcatT, bap, bag, bbp, bbg, bg, a_t, b_t, gbuf);
  tm_stage2<<<CDIM * (NDIM / 128) * (NDIM / 128), 256, 0, stream>>>(a_t, b_t, tri);
  tm_stage3<<<NNSQ / 64, 256, 0, stream>>>(tri, gbuf, lnog, lnob, WzT, bz,
                                           (float*)d_out);
}
